// GATEncoderBN_88991722373127
// MI455X (gfx1250) — matrix operand order repaired and checked by reading, not compiled, not run on hardware
//
#include <hip/hip_runtime.h>
#include <hip/hip_bf16.h>

// ---------------------------------------------------------------------------
// 2-layer GAT + BN on gfx1250 (MI455X).
// GEMMs: v_wmma_f32_16x16x32_bf16 (wave32, 16x16 tile/wave, K=128 = 4 WMMAs).
// Edge softmax+aggregation: 3 passes with f32/u32 global atomics (L2-resident:
// h[100k,128] and out[100k,128] are 51MB each, both fit in the 192MB L2).
// ---------------------------------------------------------------------------

#define D 128
#define LEAKY 0.2f
#define BN_EPSF 1e-5f

typedef __attribute__((ext_vector_type(16))) __bf16 v16bf;
typedef __attribute__((ext_vector_type(8)))  __bf16 v8bf;
typedef __attribute__((ext_vector_type(8)))  float  v8f;

// ---- float <-> totally-ordered uint (for atomicMax on floats) -------------
__device__ __forceinline__ unsigned f2ord(float f) {
  unsigned u = __float_as_uint(f);
  return (u & 0x80000000u) ? ~u : (u | 0x80000000u);
}
__device__ __forceinline__ float ord2f(unsigned u) {
  return (u & 0x80000000u) ? __uint_as_float(u & 0x7fffffffu)
                           : __uint_as_float(~u);
}

// ---- small utility kernels -------------------------------------------------
__global__ void fill_u32(unsigned* p, unsigned v, int n) {
  int i = blockIdx.x * blockDim.x + threadIdx.x;
  if (i < n) p[i] = v;
}
__global__ void fill_f32(float* p, float v, int n) {
  int i = blockIdx.x * blockDim.x + threadIdx.x;
  if (i < n) p[i] = v;
}
__global__ void init_bias(float* out, const float* __restrict__ b, int total) {
  int i = blockIdx.x * blockDim.x + threadIdx.x;
  if (i < total) out[i] = b[i & (D - 1)];
}
__global__ void relu_inplace(float* p, int n) {
  int i = blockIdx.x * blockDim.x + threadIdx.x;
  if (i < n) { float v = p[i]; p[i] = v > 0.f ? v : 0.f; }
}
__global__ void convert_x_bf16(const float* __restrict__ x, __bf16* __restrict__ xb, int n) {
  int i = blockIdx.x * blockDim.x + threadIdx.x;
  if (i < n) xb[i] = (__bf16)x[i];
}
// W[k][n] row-major f32 -> WbT[n][k] bf16 (so B fragments load contiguously)
__global__ void convert_w_bf16T(const float* __restrict__ W, __bf16* __restrict__ WbT) {
  int i = blockIdx.x * blockDim.x + threadIdx.x;  // D*D threads
  if (i < D * D) {
    int k = i >> 7, n = i & (D - 1);
    WbT[n * D + k] = (__bf16)W[k * D + n];
  }
}

// ---- WMMA GEMM: H[N,128] = A[N,128](bf16) x W[128,128] (BT = W^T bf16) ----
// 256 threads = 8 waves; wave w owns output tile (blockIdx.x*16 rows, w*16 cols)
__global__ __launch_bounds__(256)
void gemm_bf16_wmma(const __bf16* __restrict__ A, const __bf16* __restrict__ BT,
                    float* __restrict__ H, int nrows) {
  const int wave = threadIdx.x >> 5;      // 0..7 -> column tile
  const int lane = threadIdx.x & 31;
  const int half = lane >> 4;             // 0/1: which 16-lane group
  const int rbase = blockIdx.x * 16;
  int r = rbase + (lane & 15);
  int rload = r < nrows ? r : nrows - 1;  // clamp (EXEC must stay all-ones for WMMA)
  const int n = wave * 16 + (lane & 15);
  const __bf16* __restrict__ arow = A + (size_t)rload * D;
  const __bf16* __restrict__ bcol = BT + (size_t)n * D;

  v8f acc = {};
#pragma unroll
  for (int k0 = 0; k0 < D; k0 += 32) {
    // A 16x32 bf16 fragment (ISA 7.12.2): lanes0-15 rows, VGPR0-3: K=k0+half*8+0..7,
    // VGPR4-7: K=k0+16+half*8+0..7  -> two contiguous 16B loads per lane.
    v8bf a0 = *(const v8bf*)(arow + k0 + half * 8);
    v8bf a1 = *(const v8bf*)(arow + k0 + 16 + half * 8);
    v16bf a = __builtin_shufflevector(a0, a1, 0,1,2,3,4,5,6,7,8,9,10,11,12,13,14,15);
    // B 32x16 bf16 fragment: lane = column n, same K order as A:
    // VGPR0-3: K=k0+half*8+0..7, VGPR4-7: K=k0+16+half*8+0..7 (along BT[n][k]).
    v8bf b0 = *(const v8bf*)(bcol + k0 + half * 8);
    v8bf b1 = *(const v8bf*)(bcol + k0 + 16 + half * 8);
    v16bf b = __builtin_shufflevector(b0, b1, 0,1,2,3,4,5,6,7,8,9,10,11,12,13,14,15);
    acc = __builtin_amdgcn_wmma_f32_16x16x32_bf16(false, a, false, b,
                                                  (short)0, acc, false, false);
  }
  // C/D layout: VGPR v -> row = half*8 + v, col = lane&15 (within tile).
  // Bounds check is uniform per tile: hoist to one scalar branch so the
  // common (full-tile) path is 8 unguarded global_store_b32.
  float* __restrict__ hout = H + (size_t)(rbase + half * 8) * D + n;
  if (rbase + 16 <= nrows) {
#pragma unroll
    for (int v = 0; v < 8; ++v) hout[(size_t)v * D] = acc[v];
  } else {
#pragma unroll
    for (int v = 0; v < 8; ++v)
      if (rbase + half * 8 + v < nrows) hout[(size_t)v * D] = acc[v];
  }
}

// ---- per-node attention logits: as[n] = h[n].a_src, ad[n] = h[n].a_dst ----
__global__ __launch_bounds__(256)
void node_alpha(const float* __restrict__ h, const float* __restrict__ a_src,
                const float* __restrict__ a_dst, float* __restrict__ as_,
                float* __restrict__ ad_, int n) {
  int wave = threadIdx.x >> 5, lane = threadIdx.x & 31;
  int node = blockIdx.x * 8 + wave;
  if (node >= n) return;
  float4 hv = ((const float4*)(h + (size_t)node * D))[lane];
  float4 sv = ((const float4*)a_src)[lane];
  float4 dv = ((const float4*)a_dst)[lane];
  float vs = hv.x * sv.x + hv.y * sv.y + hv.z * sv.z + hv.w * sv.w;
  float vd = hv.x * dv.x + hv.y * dv.y + hv.z * dv.z + hv.w * dv.w;
#pragma unroll
  for (int m = 16; m; m >>= 1) {
    vs += __shfl_xor(vs, m, 32);
    vd += __shfl_xor(vd, m, 32);
  }
  if (lane == 0) { as_[node] = vs; ad_[node] = vd; }
}

// edge i in [0, E+N): first E from edge_index, last N are self loops
__device__ __forceinline__ void edge_sd(const int* __restrict__ ei, int E, int i,
                                        int& s, int& d) {
  if (i < E) { s = ei[i]; d = ei[E + i]; }
  else       { s = d = i - E; }
}

// ---- pass 1: e = leakyrelu(as[src]+ad[dst]); segment max over dst ---------
__global__ void edge_pass1(const int* __restrict__ ei, const float* __restrict__ as_,
                           const float* __restrict__ ad_, float* __restrict__ e,
                           unsigned* __restrict__ m_u, int E, int N) {
  int i = blockIdx.x * blockDim.x + threadIdx.x;
  if (i >= E + N) return;
  int s, d; edge_sd(ei, E, i, s, d);
  float v = as_[s] + ad_[d];
  v = v >= 0.f ? v : LEAKY * v;
  e[i] = v;
  atomicMax(m_u + d, f2ord(v));
}

// ---- pass 2: w = exp(e - m[dst]); segment sum over dst --------------------
__global__ void edge_pass2(const int* __restrict__ ei, const float* __restrict__ e,
                           const unsigned* __restrict__ m_u, float* __restrict__ w,
                           float* __restrict__ zsum, int E, int N) {
  int i = blockIdx.x * blockDim.x + threadIdx.x;
  if (i >= E + N) return;
  int s, d; edge_sd(ei, E, i, s, d);
  float wv = __expf(e[i] - ord2f(m_u[d]));
  w[i] = wv;
  atomicAdd(zsum + d, wv);
}

// ---- pass 3: out[dst] += (w/z[dst]) * h[src]  (one wave per edge) ---------
__global__ __launch_bounds__(256)
void edge_pass3(const int* __restrict__ ei, const float* __restrict__ w,
                const float* __restrict__ zsum, const float* __restrict__ h,
                float* __restrict__ out, int E, int N) {
  int wave = threadIdx.x >> 5, lane = threadIdx.x & 31;
  int i = blockIdx.x * 8 + wave;
  if (i >= E + N) return;
  int s, d; edge_sd(ei, E, i, s, d);
  float alpha = w[i] / zsum[d];
  float4 hv = ((const float4*)(h + (size_t)s * D))[lane];   // L2-resident gather
  float* op = out + (size_t)d * D + lane * 4;
  atomicAdd(op + 0, alpha * hv.x);
  atomicAdd(op + 1, alpha * hv.y);
  atomicAdd(op + 2, alpha * hv.z);
  atomicAdd(op + 3, alpha * hv.w);
}

// ---- batch-norm statistics (sum + sumsq per channel) ----------------------
__global__ __launch_bounds__(256)
void bn_stats(const float* __restrict__ z, float* __restrict__ sum,
              float* __restrict__ sq, int n) {
  __shared__ float ls[D], lq[D];
  int c = threadIdx.x & (D - 1);
  int half = threadIdx.x >> 7;
  float s = 0.f, q = 0.f;
  for (int r = blockIdx.x * 2 + half; r < n; r += gridDim.x * 2) {
    float v = z[(size_t)r * D + c];
    s += v; q += v * v;
  }
  if (half) { ls[c] = s; lq[c] = q; }
  __syncthreads();
  if (!half) { atomicAdd(sum + c, s + ls[c]); atomicAdd(sq + c, q + lq[c]); }
}

// ---- fused BN + ReLU + bf16 convert (feeds layer-2 WMMA GEMM) -------------
__global__ void bn_apply_relu_bf16(const float* __restrict__ z, const float* __restrict__ sum,
                                   const float* __restrict__ sq, const float* __restrict__ gamma,
                                   const float* __restrict__ beta, __bf16* __restrict__ xb, int n) {
  int i = blockIdx.x * blockDim.x + threadIdx.x;
  if (i >= n * D) return;
  int c = i & (D - 1);
  float inv_n = 1.f / (float)n;
  float mu = sum[c] * inv_n;
  float var = sq[c] * inv_n - mu * mu;
  float sc = rsqrtf(var + BN_EPSF) * gamma[c];
  float v = (z[i] - mu) * sc + beta[c];
  v = v > 0.f ? v : 0.f;
  xb[i] = (__bf16)v;
}

// ---------------------------------------------------------------------------
extern "C" void kernel_launch(void* const* d_in, const int* in_sizes, int n_in,
                              void* d_out, int out_size, void* d_ws, size_t ws_size,
                              hipStream_t stream) {
  const float* x      = (const float*)d_in[0];
  const int*   ei     = (const int*)d_in[1];
  const float* W1     = (const float*)d_in[2];
  const float* a1_src = (const float*)d_in[3];
  const float* a1_dst = (const float*)d_in[4];
  const float* b1     = (const float*)d_in[5];
  const float* gamma  = (const float*)d_in[6];
  const float* beta   = (const float*)d_in[7];
  const float* W2     = (const float*)d_in[8];
  const float* a2_src = (const float*)d_in[9];
  const float* a2_dst = (const float*)d_in[10];
  const float* b2     = (const float*)d_in[11];

  const int N  = in_sizes[0] / D;
  const int E  = in_sizes[1] / 2;
  const int ET = E + N;                 // with self loops

  // ---- carve workspace ----
  size_t off = 0;
  auto carve = [&](size_t bytes) -> void* {
    void* p = (char*)d_ws + off;
    off += (bytes + 255) & ~(size_t)255;
    return p;
  };
  float*    h     = (float*)carve((size_t)N * D * 4);
  __bf16*   xb    = (__bf16*)carve((size_t)N * D * 2);
  __bf16*   WbT   = (__bf16*)carve((size_t)D * D * 2);
  float*    as_   = (float*)carve((size_t)N * 4);
  float*    ad_   = (float*)carve((size_t)N * 4);
  unsigned* m_u   = (unsigned*)carve((size_t)N * 4);
  float*    zsum  = (float*)carve((size_t)N * 4);
  float*    ebuf  = (float*)carve((size_t)ET * 4);
  float*    wbuf  = (float*)carve((size_t)ET * 4);
  float*    z1    = (float*)carve((size_t)N * D * 4);
  float*    bnsum = (float*)carve(2 * D * 4);
  float*    bnsq  = bnsum + D;

  const int TPB = 256;
  const int gElem = (N * D + TPB - 1) / TPB;   // N*D elementwise
  const int gEdge = (ET + TPB - 1) / TPB;      // per-edge, 1 thread/edge
  const int gEdgeW = (ET + 7) / 8;             // per-edge, 1 wave/edge
  const int gRow16 = (N + 15) / 16;            // GEMM row tiles
  const int gNodeW = (N + 7) / 8;              // 1 wave/node
  const unsigned NEG_INF_ORD = 0x007FFFFFu;    // f2ord(-inf)

  auto run_layer = [&](const __bf16* xin, const float* W, const float* a_s,
                       const float* a_d, const float* bias, float* out_f32) {
    convert_w_bf16T<<<(D * D + TPB - 1) / TPB, TPB, 0, stream>>>(W, WbT);
    gemm_bf16_wmma<<<gRow16, TPB, 0, stream>>>(xin, WbT, h, N);
    node_alpha<<<gNodeW, TPB, 0, stream>>>(h, a_s, a_d, as_, ad_, N);
    fill_u32<<<(N + TPB - 1) / TPB, TPB, 0, stream>>>(m_u, NEG_INF_ORD, N);
    fill_f32<<<(N + TPB - 1) / TPB, TPB, 0, stream>>>(zsum, 0.f, N);
    init_bias<<<gElem, TPB, 0, stream>>>(out_f32, bias, N * D);
    edge_pass1<<<gEdge, TPB, 0, stream>>>(ei, as_, ad_, ebuf, m_u, E, N);
    edge_pass2<<<gEdge, TPB, 0, stream>>>(ei, ebuf, m_u, wbuf, zsum, E, N);
    edge_pass3<<<gEdgeW, TPB, 0, stream>>>(ei, wbuf, zsum, h, out_f32, E, N);
  };

  // layer 1
  convert_x_bf16<<<gElem, TPB, 0, stream>>>(x, xb, N * D);
  run_layer(xb, W1, a1_src, a1_dst, b1, z1);

  // batchnorm + relu -> bf16 input for layer 2
  fill_f32<<<1, 2 * D, 0, stream>>>(bnsum, 0.f, 2 * D);
  bn_stats<<<1024, TPB, 0, stream>>>(z1, bnsum, bnsq, N);
  bn_apply_relu_bf16<<<gElem, TPB, 0, stream>>>(z1, bnsum, bnsq, gamma, beta, xb, N);

  // layer 2 (accumulate straight into d_out) + final relu
  run_layer(xb, W2, a2_src, a2_dst, b2, (float*)d_out);
  relu_inplace<<<(out_size + TPB - 1) / TPB, TPB, 0, stream>>>((float*)d_out, out_size);

  (void)n_in; (void)ws_size;
}
